// FalconAttention_21732534518106
// MI455X (gfx1250) — matrix operand order repaired and checked by reading, not compiled, not run on hardware
//
#include <hip/hip_runtime.h>
#include <stdint.h>

// Problem constants (from reference)
#define B_   2
#define S_   2048
#define D_   4096
#define H_   32
#define KVH_ 8
#define HD_  128
#define E_   6144          // (KVH*2+H)*HD
#define M_   (B_*S_)       // 4096 tokens

typedef __attribute__((ext_vector_type(16))) __bf16 v16bf;
typedef __attribute__((ext_vector_type(8)))  float  v8f;

union Frag32B { uint4 u[2]; v16bf v; };

__device__ __forceinline__ uint16_t f2bf(float f){
  uint32_t u = __float_as_uint(f);
  u += 0x7FFFu + ((u >> 16) & 1u);          // round-to-nearest-even
  return (uint16_t)(u >> 16);
}
__device__ __forceinline__ float bf2f(uint16_t h){
  return __uint_as_float(((uint32_t)h) << 16);
}

// A-fragment (16x32 bf16): lane holds 8 contiguous elems @p and 8 @p+16
__device__ __forceinline__ v16bf ldA(const uint16_t* p){
  Frag32B f; f.u[0] = *(const uint4*)p; f.u[1] = *(const uint4*)(p + 16); return f.v;
}
// B-fragment (32x16 bf16): lane holds 8 contiguous K elems @p and 8 @p+16
__device__ __forceinline__ v16bf ldB(const uint16_t* p){
  Frag32B f; f.u[0] = *(const uint4*)p; f.u[1] = *(const uint4*)(p + 16); return f.v;
}
#define WMMA_BF16(a,b,c) \
  __builtin_amdgcn_wmma_f32_16x16x32_bf16(false,(a),false,(b),(short)0,(c),false,false)

// ---- CDNA5 async-to-LDS copy (ASYNCcnt path), inline asm (assembles on gfx1250).
__device__ __forceinline__ void async_b128(uint32_t lds_off, const uint16_t* g){
  asm volatile("global_load_async_to_lds_b128 %0, %1, off"
               :: "v"(lds_off), "v"(g) : "memory");
}
__device__ __forceinline__ void wait_async0(){
  asm volatile("s_wait_asynccnt 0x0" ::: "memory");
}
__device__ __forceinline__ uint32_t lds_off(const uint16_t* p){
  return (uint32_t)(uintptr_t)p;   // LDS aperture: addr[31:0] is the LDS byte offset
}

// ---------------------------------------------------------------- convert f32 -> bf16
__global__ void cvt_bf16(const float* __restrict__ src, uint16_t* __restrict__ dst, long n){
  long i = ((long)blockIdx.x*blockDim.x + threadIdx.x)*4;
  const long stride = (long)gridDim.x*blockDim.x*4;
  for(; i < n; i += stride){
    float4 f = *(const float4*)(src + i);
    uint2 o;
    o.x = (uint32_t)f2bf(f.x) | ((uint32_t)f2bf(f.y) << 16);
    o.y = (uint32_t)f2bf(f.z) | ((uint32_t)f2bf(f.w) << 16);
    *(uint2*)(dst + i) = o;
  }
}

// ---------------------------------------------------------------- bf16 GEMM, C = A[M,K] * W[N,K]^T
// Block = 256 threads / 8 waves, block tile 128(M) x 128(N), k-step 32.
// Double-buffered LDS tiles filled with global_load_async_to_lds_b128.
#define LDS_STRIDE 40

__device__ __forceinline__ void prefetch_tiles(const uint16_t* A, const uint16_t* W, int K,
                                               int m0, int n0, int k0,
                                               uint16_t* sA, uint16_t* sB, int tid){
  #pragma unroll
  for(int q = 0; q < 2; q++){
    const int c   = tid + q*256;     // chunk id 0..511
    const int row = c >> 2;          // 0..127
    const int kc  = c & 3;           // 16B chunk within the 32-elem k slice
    async_b128(lds_off(sA + row*LDS_STRIDE + kc*8), A + (size_t)(m0 + row)*K + k0 + kc*8);
  }
  #pragma unroll
  for(int q = 0; q < 2; q++){
    const int c   = tid + q*256;
    const int row = c >> 2;
    const int kc  = c & 3;
    async_b128(lds_off(sB + row*LDS_STRIDE + kc*8), W + (size_t)(n0 + row)*K + k0 + kc*8);
  }
}

template<bool OUT_F32>
__global__ __launch_bounds__(256) void gemm_bf16(const uint16_t* __restrict__ A,
                                                 const uint16_t* __restrict__ W,
                                                 void* __restrict__ C,
                                                 int M, int N, int K)
{
  __shared__ uint16_t sA[2][128*LDS_STRIDE];   // 2 x 10 KB
  __shared__ uint16_t sB[2][128*LDS_STRIDE];   // 2 x 10 KB
  const int tid  = threadIdx.x;
  const int lane = tid & 31;
  const int wave = tid >> 5;
  const int wm   = wave >> 1;          // 0..3 : 32-row M band
  const int wn   = wave & 1;           // 0..1 : 64-col N band
  const int m0   = blockIdx.y*128;
  const int n0   = blockIdx.x*128;
  const int half = lane >> 4;
  const int ln16 = lane & 15;

  v8f acc[2][4];
  #pragma unroll
  for(int s = 0; s < 2; s++)
    #pragma unroll
    for(int t = 0; t < 4; t++) acc[s][t] = 0;

  prefetch_tiles(A, W, K, m0, n0, 0, sA[0], sB[0], tid);

  const int NK = K / 32;
  for(int kt = 0; kt < NK; kt++){
    const int cur = kt & 1;
    wait_async0();                 // our slice of tile `cur` has landed in LDS
    __syncthreads();               // whole tile visible
    if (kt + 1 < NK)
      prefetch_tiles(A, W, K, m0, n0, (kt+1)*32, sA[cur^1], sB[cur^1], tid);  // overlaps compute

    const uint16_t* a_base = &sA[cur][(wm*32 + ln16)*LDS_STRIDE + half*8];
    const v16bf fa0 = ldA(a_base);                     // rows wm*32 + 0..15
    const v16bf fa1 = ldA(a_base + 16*LDS_STRIDE);     // rows wm*32 + 16..31
    #pragma unroll
    for(int t = 0; t < 4; t++){
      const v16bf fb = ldB(&sB[cur][(wn*64 + t*16 + ln16)*LDS_STRIDE + half*8]);
      acc[0][t] = WMMA_BF16(fa0, fb, acc[0][t]);
      acc[1][t] = WMMA_BF16(fa1, fb, acc[1][t]);
    }
    __syncthreads();               // all waves done reading tile `cur` before refill
  }

  const int colb = n0 + wn*64 + ln16;
  #pragma unroll
  for(int sub = 0; sub < 2; sub++){
    const int rowb = m0 + wm*32 + sub*16 + half*8;
    #pragma unroll
    for(int t = 0; t < 4; t++)
      #pragma unroll
      for(int i = 0; i < 8; i++){
        const size_t idx = (size_t)(rowb + i)*N + colb + t*16;
        if (OUT_F32) ((float*)C)[idx]    = acc[sub][t][i];
        else         ((uint16_t*)C)[idx] = f2bf(acc[sub][t][i]);
      }
  }
}

// ---------------------------------------------------------------- RoPE + scatter fused QKV
__global__ __launch_bounds__(128) void rope_scatter(const uint16_t* __restrict__ Fused,
                                                    uint16_t* __restrict__ Qo,   // [B,H,S,HD]
                                                    uint16_t* __restrict__ Ko,   // [B,KVH,S,HD]
                                                    uint16_t* __restrict__ Vt)   // [B,KVH,HD,S]
{
  const int hd  = threadIdx.x;
  const int s   = blockIdx.x;
  const int b   = blockIdx.y;
  const int kvh = blockIdx.z;
  const uint16_t* base = Fused + ((size_t)(b*S_ + s))*E_ + kvh*768;
  const int j = hd & 63;
  const float invf = __expf(-((float)(2*j) / 128.0f) * 9.210340371976184f); // ln(10000)
  const float ang = (float)s * invf;
  const float cs = __cosf(ang), sn = __sinf(ang);
  const float sgn = (hd < 64) ? -1.0f : 1.0f;
  const int partner = (hd < 64) ? hd + 64 : hd - 64;
  #pragma unroll
  for(int slot = 0; slot < 4; slot++){
    float x  = bf2f(base[slot*128 + hd]);
    float xp = bf2f(base[slot*128 + partner]);
    float val = x*cs + sgn*xp*sn;
    Qo[(((size_t)(b*H_ + kvh*4 + slot))*S_ + s)*HD_ + hd] = f2bf(val);
  }
  { // K with RoPE
    float x  = bf2f(base[4*128 + hd]);
    float xp = bf2f(base[4*128 + partner]);
    float val = x*cs + sgn*xp*sn;
    Ko[(((size_t)(b*KVH_ + kvh))*S_ + s)*HD_ + hd] = f2bf(val);
  }
  { // V transposed [hd][s] so P*V B-fragments are contiguous
    float x = bf2f(base[5*128 + hd]);
    Vt[(((size_t)(b*KVH_ + kvh))*HD_ + hd)*S_ + s] = f2bf(x);
  }
}

// ---------------------------------------------------------------- flash attention (causal, GQA)
// Block = 8 waves sharing one head; 128 queries per block (16 per wave).
// K/V tiles (32 keys) staged in double-buffered LDS via async-to-LDS, shared by all waves.
#define KSTR 136   // K-tile row stride (elems): 272B -> conflict-free frag reads
#define VSTR 40    // V-tile row stride (elems):  80B -> conflict-free frag reads

__device__ __forceinline__ void attn_prefetch(const uint16_t* kbase, const uint16_t* vbase,
                                              int kb, uint16_t* sK, uint16_t* sV, int tid){
  #pragma unroll
  for(int q = 0; q < 2; q++){
    const int c   = tid + q*256;      // 0..511
    const int row = c >> 4;           // 0..31 (key)
    const int kc  = c & 15;           // 16 chunks of 8 elems across HD=128
    async_b128(lds_off(sK + row*KSTR + kc*8), kbase + (size_t)(kb + row)*HD_ + kc*8);
  }
  #pragma unroll
  for(int q = 0; q < 2; q++){
    const int c   = tid + q*256;
    const int row = c >> 2;           // 0..127 (hd)
    const int kc  = c & 3;            // 4 chunks of 8 elems across 32 keys
    async_b128(lds_off(sV + row*VSTR + kc*8), vbase + (size_t)row*S_ + kb + kc*8);
  }
}

__global__ __launch_bounds__(256) void attn_flash(const uint16_t* __restrict__ Q,
                                                  const uint16_t* __restrict__ Kc,
                                                  const uint16_t* __restrict__ Vt,
                                                  uint16_t* __restrict__ Ctx)  // [B,S,H*HD]
{
  __shared__ uint16_t sK[2][32*KSTR];    // 2 x 8.5 KB
  __shared__ uint16_t sV[2][128*VSTR];   // 2 x 10 KB
  __shared__ uint16_t ldsP[8][16*32];    // per-wave P staging (C-layout -> A-layout)
  const int tid  = threadIdx.x;
  const int lane = tid & 31;
  const int wave = tid >> 5;
  const int b   = blockIdx.y >> 5;
  const int h   = blockIdx.y & 31;
  const int kvh = h >> 2;
  const int q0  = blockIdx.x*128 + wave*16;
  const int kmax = blockIdx.x*128 + 128;     // keys needed by the whole block
  const int half = lane >> 4;
  const int ln16 = lane & 15;
  const uint16_t* qb  = Q  + (((size_t)(b*H_ + h))*S_ + q0)*HD_;
  const uint16_t* kb_ = Kc + ((size_t)(b*KVH_ + kvh))*S_*HD_;
  const uint16_t* vb  = Vt + ((size_t)(b*KVH_ + kvh))*HD_*S_;

  v16bf qf[4];
  #pragma unroll
  for(int c = 0; c < 4; c++)
    qf[c] = ldA(qb + (size_t)ln16*HD_ + c*32 + (half << 3));

  float m[8], l[8];
  v8f acc[8];
  #pragma unroll
  for(int i = 0; i < 8; i++){ m[i] = -3.0e38f; l[i] = 0.0f; }
  #pragma unroll
  for(int t = 0; t < 8; t++) acc[t] = 0;

  const float scale = 0.08838834764831845f;  // 1/sqrt(128)
  const int rbase = q0 + (half << 3);

  attn_prefetch(kb_, vb, 0, sK[0], sV[0], tid);

  for(int kb = 0, it = 0; kb < kmax; kb += 32, it++){
    const int cur = it & 1;
    wait_async0();                 // our slice of K/V tile `cur` landed
    __syncthreads();               // whole tile visible to all waves
    if (kb + 32 < kmax)
      attn_prefetch(kb_, vb, kb + 32, sK[cur^1], sV[cur^1], tid);   // overlaps compute

    if (kb < q0 + 16) {            // wave-uniform causal skip; EXEC all-ones inside
      v8f s0 = 0, s1 = 0;
      #pragma unroll
      for(int c = 0; c < 4; c++){
        const uint16_t* p = &sK[cur][ln16*KSTR + c*32 + (half << 3)];
        s0 = WMMA_BF16(qf[c], ldB(p),           s0);
        s1 = WMMA_BF16(qf[c], ldB(p + 16*KSTR), s1);
      }
      const int key0 = kb + ln16;
      const int key1 = key0 + 16;
      #pragma unroll
      for(int i = 0; i < 8; i++){
        const int qrow = rbase + i;
        float v0 = (key0 <= qrow) ? s0[i]*scale : -3.0e38f;
        float v1 = (key1 <= qrow) ? s1[i]*scale : -3.0e38f;
        float mx = fmaxf(v0, v1);
        mx = fmaxf(mx, __shfl_xor(mx, 1, 32));
        mx = fmaxf(mx, __shfl_xor(mx, 2, 32));
        mx = fmaxf(mx, __shfl_xor(mx, 4, 32));
        mx = fmaxf(mx, __shfl_xor(mx, 8, 32));
        const float mn    = fmaxf(m[i], mx);
        const float alpha = __expf(m[i] - mn);
        const float p0 = (key0 <= qrow) ? __expf(v0 - mn) : 0.0f;
        const float p1 = (key1 <= qrow) ? __expf(v1 - mn) : 0.0f;
        float rs = p0 + p1;
        rs += __shfl_xor(rs, 1, 32);
        rs += __shfl_xor(rs, 2, 32);
        rs += __shfl_xor(rs, 4, 32);
        rs += __shfl_xor(rs, 8, 32);
        l[i] = l[i]*alpha + rs;
        m[i] = mn;
        #pragma unroll
        for(int t = 0; t < 8; t++) acc[t][i] *= alpha;
        const int r = (half << 3) + i;
        ldsP[wave][r*32 + ln16]      = f2bf(p0);
        ldsP[wave][r*32 + ln16 + 16] = f2bf(p1);
      }
      asm volatile("" ::: "memory");   // order per-wave LDS b16 stores before b128 reload
      const v16bf pf = ldA(&ldsP[wave][ln16*32 + (half << 3)]);
      #pragma unroll
      for(int t = 0; t < 8; t++){
        const v16bf fv = ldB(&sV[cur][(t*16 + ln16)*VSTR + (half << 3)]);
        acc[t] = WMMA_BF16(pf, fv, acc[t]);
      }
    }
    __syncthreads();               // all waves done reading tile `cur` before refill
  }

  #pragma unroll
  for(int i = 0; i < 8; i++){
    const float inv = 1.0f / l[i];
    const size_t rowb = ((size_t)(b*S_ + rbase + i))*(H_*HD_) + (size_t)h*HD_ + ln16;
    #pragma unroll
    for(int t = 0; t < 8; t++)
      Ctx[rowb + t*16] = f2bf(acc[t][i] * inv);
  }
}

// ---------------------------------------------------------------- launcher
extern "C" void kernel_launch(void* const* d_in, const int* in_sizes, int n_in,
                              void* d_out, int out_size, void* d_ws, size_t ws_size,
                              hipStream_t stream)
{
  const float* hidden  = (const float*)d_in[0];   // [B,S,D]
  const float* w_qkv   = (const float*)d_in[1];   // [E,D]
  const float* w_dense = (const float*)d_in[2];   // [D,D]
  char* ws = (char*)d_ws;

  // workspace layout (bytes), all 256B aligned; total ~252 MB
  uint16_t* hid_bf   = (uint16_t*)(ws + 0);          // 32 MB
  uint16_t* wqkv_bf  = (uint16_t*)(ws + 33554432);   // 48 MB
  uint16_t* wden_bf  = (uint16_t*)(ws + 83886080);   // 32 MB
  uint16_t* fused_bf = (uint16_t*)(ws + 117440512);  // 48 MB  [M, E]
  uint16_t* q_bf     = (uint16_t*)(ws + 167772160);  // 32 MB  [B,H,S,HD]
  uint16_t* k_bf     = (uint16_t*)(ws + 201326592);  //  8 MB  [B,KVH,S,HD]
  uint16_t* vt_bf    = (uint16_t*)(ws + 209715200);  //  8 MB  [B,KVH,HD,S]
  uint16_t* ctx_bf   = (uint16_t*)(ws + 218103808);  // 32 MB  [M, H*HD]

  cvt_bf16<<<2048, 256, 0, stream>>>(hidden,  hid_bf,  (long)M_*D_);
  cvt_bf16<<<2048, 256, 0, stream>>>(w_qkv,   wqkv_bf, (long)E_*D_);
  cvt_bf16<<<2048, 256, 0, stream>>>(w_dense, wden_bf, (long)D_*D_);

  // QKV projection: [4096,4096] x [6144,4096]^T -> bf16 fused
  gemm_bf16<false><<<dim3(E_/128, M_/128), 256, 0, stream>>>(hid_bf, wqkv_bf, fused_bf, M_, E_, D_);

  rope_scatter<<<dim3(S_, B_, KVH_), 128, 0, stream>>>(fused_bf, q_bf, k_bf, vt_bf);

  attn_flash<<<dim3(S_/128, B_*H_), 256, 0, stream>>>(q_bf, k_bf, vt_bf, ctx_bf);

  // dense projection: ctx[4096,4096] x w_dense[4096,4096]^T -> f32 d_out
  gemm_bf16<true><<<dim3(D_/128, M_/128), 256, 0, stream>>>(ctx_bf, wden_bf, d_out, M_, D_, D_);
}
